// BilateralLSTMCell_23012434772050
// MI455X (gfx1250) — matrix operand order repaired and checked by reading, not compiled, not run on hardware
//
#include <hip/hip_runtime.h>
#include <hip/hip_bf16.h>

#define BATCH 4096
#define DIN   512
#define HDIM  1024
#define KTOT  2560   // 512 + 1024 + 1024
#define KSTEPS (KTOT / 32)

typedef __attribute__((ext_vector_type(16))) __bf16        v16bf;
typedef __attribute__((ext_vector_type(8)))  float         v8f;
typedef __attribute__((ext_vector_type(4)))  unsigned int  u32x4;
typedef __attribute__((ext_vector_type(2)))  unsigned int  u32x2;
typedef __attribute__((ext_vector_type(4)))  float         f32x4;

union FragBF {
    v16bf v;
    u32x4 q[2];
};

static __device__ __forceinline__ unsigned short f2bf(float f) {
    unsigned int u = __float_as_uint(f);
    // round-to-nearest-even
    unsigned int r = (u + 0x7FFFu + ((u >> 16) & 1u)) >> 16;
    return (unsigned short)r;
}

// ---------------------------------------------------------------------------
// Kernel 1: build A_bf16 [4096 x 2560] = [x | h_prev | h_cross], row-major.
// ---------------------------------------------------------------------------
__global__ __launch_bounds__(256) void pack_A(const float* __restrict__ x,
                                              const float* __restrict__ hm1,
                                              const float* __restrict__ hm2,
                                              unsigned short* __restrict__ Abf) {
    size_t idx = ((size_t)blockIdx.x * blockDim.x + threadIdx.x) * 4;
    if (idx >= (size_t)BATCH * KTOT) return;
    int m = (int)(idx / KTOT);
    int k = (int)(idx % KTOT);
    const float* src;
    if (k < DIN)              src = x   + (size_t)m * DIN       + k;
    else if (k < DIN + HDIM)  src = hm1 + (size_t)m * (2*HDIM)  + (k - DIN);
    else                      src = hm2 + (size_t)m * (2*HDIM)  + (k - DIN - HDIM);
    f32x4 v = *(const f32x4*)src;
    u32x2 o;
    o.x = (unsigned int)f2bf(v.x) | ((unsigned int)f2bf(v.y) << 16);
    o.y = (unsigned int)f2bf(v.z) | ((unsigned int)f2bf(v.w) << 16);
    *(u32x2*)(Abf + idx) = o;
}

// ---------------------------------------------------------------------------
// Kernel 2: build Wt_bf16 [gate][n=1024][k=2560] (K-major), LDS-tiled transpose.
// ---------------------------------------------------------------------------
__global__ __launch_bounds__(256) void pack_Wt(
    const float* __restrict__ Wi, const float* __restrict__ Wf,
    const float* __restrict__ Wo, const float* __restrict__ Wc,
    const float* __restrict__ Ui, const float* __restrict__ Uf,
    const float* __restrict__ Uo, const float* __restrict__ Uc,
    const float* __restrict__ Vi, const float* __restrict__ Vf,
    const float* __restrict__ Vo, const float* __restrict__ Vc,
    unsigned short* __restrict__ Wt) {
    __shared__ unsigned short tile[32][33];
    int g = blockIdx.z;
    const float* Wg = (g == 0) ? Wi : (g == 1) ? Wf : (g == 2) ? Wo : Wc;
    const float* Ug = (g == 0) ? Ui : (g == 1) ? Uf : (g == 2) ? Uo : Uc;
    const float* Vg = (g == 0) ? Vi : (g == 1) ? Vf : (g == 2) ? Vo : Vc;
    int k00 = blockIdx.x * 32;
    int n00 = blockIdx.y * 32;
    int tx = threadIdx.x;   // 0..31
    int ty = threadIdx.y;   // 0..7
#pragma unroll
    for (int i = 0; i < 4; ++i) {
        int k = k00 + ty + i * 8;
        int n = n00 + tx;
        const float* src; int kk;
        if (k < DIN)             { src = Wg; kk = k; }
        else if (k < DIN + HDIM) { src = Ug; kk = k - DIN; }
        else                     { src = Vg; kk = k - DIN - HDIM; }
        tile[ty + i * 8][tx] = f2bf(src[(size_t)kk * HDIM + n]);
    }
    __syncthreads();
#pragma unroll
    for (int i = 0; i < 4; ++i) {
        int n = n00 + ty + i * 8;
        int k = k00 + tx;
        Wt[((size_t)g * HDIM + n) * KTOT + k] = tile[tx][ty + i * 8];
    }
}

// ---------------------------------------------------------------------------
// Kernel 3: fused GEMM (bf16 WMMA, f32 accum) + LSTM epilogue.
// Block: 256 threads = 8 waves arranged 2(M) x 4(N).
// Block tile: 128 rows x 64 cols per gate (all 4 gates resident).
// Wave tile:  64 x 16 per gate -> 4 M-tiles x 4 gates = 16 v8f accumulators.
// Double-buffered LDS staging (60 KB of the 320 KB/WGP).
// K loop kept rolled (#pragma unroll 1) and B fragments cached / A streamed
// to keep the live set < 256 VGPRs (no s_set_vgpr_msb / acc-move overhead).
// ---------------------------------------------------------------------------
__global__ __launch_bounds__(256) void gemm_lstm(
    const unsigned short* __restrict__ Abf,
    const unsigned short* __restrict__ Wt,
    const float* __restrict__ hm1,
    float* __restrict__ out) {
    __shared__ __align__(16) unsigned short As[2][128][40];
    __shared__ __align__(16) unsigned short Bs[2][4][64][40];

    const int t    = threadIdx.x;
    const int wave = t >> 5;
    const int lane = t & 31;
    const int wm   = wave & 1;        // 0..1 -> 64 rows each
    const int wn   = wave >> 1;       // 0..3 -> 16 cols each (per gate)
    const int m0b  = blockIdx.x * 128;
    const int n0b  = blockIdx.y * 64;
    const int half = lane >> 4;
    const int l16  = lane & 15;

    // copy-thread mapping: A tile 128x32 (2 b128/thread), B tile 4x64x32 (4 b128/thread)
    const int arow = t >> 1;          // 0..127
    const int akh  = t & 1;           // which 16-elem half of the 32-wide k slab
    const int bg   = t >> 6;          // 0..3 gate
    const int bn   = t & 63;          // 0..63 col
    const unsigned short* Agp = Abf + (size_t)(m0b + arow) * KTOT + akh * 16;
    const unsigned short* Bgp = Wt  + ((size_t)bg * HDIM + n0b + bn) * KTOT;

    v8f acc[4][4] = {};               // [gate][mtile]

    // prologue: global loads for k-tile 0
    u32x4 ra0 = *(const u32x4*)(Agp + 0);
    u32x4 ra1 = *(const u32x4*)(Agp + 8);
    u32x4 rb0 = *(const u32x4*)(Bgp + 0);
    u32x4 rb1 = *(const u32x4*)(Bgp + 8);
    u32x4 rb2 = *(const u32x4*)(Bgp + 16);
    u32x4 rb3 = *(const u32x4*)(Bgp + 24);

#pragma unroll 1
    for (int kt = 0; kt < KSTEPS; ++kt) {
        const int buf = kt & 1;
        // stage current k-tile into LDS
        *(u32x4*)&As[buf][arow][akh * 16]     = ra0;
        *(u32x4*)&As[buf][arow][akh * 16 + 8] = ra1;
        *(u32x4*)&Bs[buf][bg][bn][0]          = rb0;
        *(u32x4*)&Bs[buf][bg][bn][8]          = rb1;
        *(u32x4*)&Bs[buf][bg][bn][16]         = rb2;
        *(u32x4*)&Bs[buf][bg][bn][24]         = rb3;
        __syncthreads();
        // prefetch next k-tile from global (L2-resident); overlaps compute below
        if (kt + 1 < KSTEPS) {
            const int k0 = (kt + 1) * 32;
            ra0 = *(const u32x4*)(Agp + k0);
            ra1 = *(const u32x4*)(Agp + k0 + 8);
            rb0 = *(const u32x4*)(Bgp + k0);
            rb1 = *(const u32x4*)(Bgp + k0 + 8);
            rb2 = *(const u32x4*)(Bgp + k0 + 16);
            rb3 = *(const u32x4*)(Bgp + k0 + 24);
        }
        // B fragments (32x16): lanes 0-15 K 0-15, lanes 16-31 K 16-31;
        // K-major Bs -> 32 contiguous bytes at elem half*16. Cache all 4 gates.
        FragBF b[4];
#pragma unroll
        for (int g = 0; g < 4; ++g) {
            const unsigned short* bp = &Bs[buf][g][wn * 16 + l16][half * 16];
            b[g].q[0] = *(const u32x4*)bp;
            b[g].q[1] = *(const u32x4*)(bp + 8);
        }
        // A fragments streamed one at a time (contiguous k order, same as B):
        // lanes 0-15 K 0-15, lanes 16-31 K 16-31.
#pragma unroll
        for (int mt = 0; mt < 4; ++mt) {
            const unsigned short* ap = &As[buf][wm * 64 + mt * 16 + l16][half * 16];
            FragBF a;
            a.q[0] = *(const u32x4*)ap;
            a.q[1] = *(const u32x4*)(ap + 8);
#pragma unroll
            for (int g = 0; g < 4; ++g)
                acc[g][mt] = __builtin_amdgcn_wmma_f32_16x16x32_bf16(
                    false, a.v, false, b[g].v, (short)0, acc[g][mt], false, false);
        }
    }

    // Epilogue: C/D layout — lanes 0-15: M=r, N=lane; lanes 16-31: M=8+r.
    float* h_out = out;                               // [4096 x 1024]
    float* ns    = out + (size_t)BATCH * HDIM;        // new_state [4096 x 2048]
    const int n = n0b + wn * 16 + l16;
#pragma unroll
    for (int mt = 0; mt < 4; ++mt) {
#pragma unroll
        for (int r = 0; r < 8; ++r) {
            int m = m0b + wm * 64 + mt * 16 + half * 8 + r;
            float gi = acc[0][mt][r];
            float gf = acc[1][mt][r];
            float go = acc[2][mt][r];
            float gc = acc[3][mt][r];
            float i_ = 1.0f / (1.0f + __expf(-gi));
            float f_ = 1.0f / (1.0f + __expf(-gf));
            float o_ = 1.0f / (1.0f + __expf(-go));
            float cc = tanhf(gc);
            float cp = hm1[(size_t)m * (2 * HDIM) + HDIM + n];
            float c  = f_ * cp + i_ * cc;
            float h  = o_ * tanhf(c);
            h_out[(size_t)m * HDIM + n]           = h;
            ns[(size_t)m * (2 * HDIM) + n]        = h;
            ns[(size_t)m * (2 * HDIM) + HDIM + n] = c;
        }
    }
}

// ---------------------------------------------------------------------------
extern "C" void kernel_launch(void* const* d_in, const int* in_sizes, int n_in,
                              void* d_out, int out_size, void* d_ws, size_t ws_size,
                              hipStream_t stream) {
    const float* x   = (const float*)d_in[0];
    const float* hm1 = (const float*)d_in[1];
    const float* hm2 = (const float*)d_in[2];
    // setup_inputs order: Wi,Wf,Wo,Wc (3..6), Ui,Uf,Uo,Uc (7..10), Vi,Vf,Vo,Vc (11..14)

    unsigned short* Abf = (unsigned short*)d_ws;                 // 4096*2560 bf16 = 20 MB
    unsigned short* Wt  = Abf + (size_t)BATCH * KTOT;            // 4*1024*2560 bf16 = 20 MB

    {
        size_t total = (size_t)BATCH * KTOT / 4;
        int blocks = (int)((total + 255) / 256);
        pack_A<<<blocks, 256, 0, stream>>>(x, hm1, hm2, Abf);
    }
    pack_Wt<<<dim3(KTOT / 32, HDIM / 32, 4), dim3(32, 8), 0, stream>>>(
        (const float*)d_in[3],  (const float*)d_in[4],
        (const float*)d_in[5],  (const float*)d_in[6],
        (const float*)d_in[7],  (const float*)d_in[8],
        (const float*)d_in[9],  (const float*)d_in[10],
        (const float*)d_in[11], (const float*)d_in[12],
        (const float*)d_in[13], (const float*)d_in[14],
        Wt);
    gemm_lstm<<<dim3(BATCH / 128, HDIM / 64), 256, 0, stream>>>(
        Abf, Wt, hm1, (float*)d_out);
}
